// MultiheadAttention_1340029796536
// MI455X (gfx1250) — matrix operand order repaired and checked by reading, not compiled, not run on hardware
//
#include <hip/hip_runtime.h>
#include <hip/hip_bf16.h>

#define D_MODEL 2048
#define NUM_HEADS 16
#define D_HEAD 128
#define SEQ 2048
#define BATCH 4
#define M_TOTAL (BATCH * SEQ)   /* 8192 */
#define N_PROJ (3 * D_MODEL)    /* 6144 */
#define SCALE_Q 0.08838834764831845f /* 1/sqrt(128) */
#define LOG2E 1.4426950408889634f

typedef __attribute__((ext_vector_type(8)))  __bf16 v8bf;
typedef __attribute__((ext_vector_type(16))) __bf16 v16bf;
typedef __attribute__((ext_vector_type(8)))  float  v8f;
typedef __attribute__((ext_vector_type(4)))  int    v4i;

__device__ __forceinline__ unsigned short f2bf(float f) {
  unsigned int u = __float_as_uint(f);
  u += 0x7FFFu + ((u >> 16) & 1u);   // round-to-nearest-even
  return (unsigned short)(u >> 16);
}
__device__ __forceinline__ v8bf  ld8 (const unsigned short* p) { return *reinterpret_cast<const v8bf *>(p); }
__device__ __forceinline__ v16bf ld16(const unsigned short* p) { return *reinterpret_cast<const v16bf*>(p); }
__device__ __forceinline__ v16bf cat8(v8bf lo, v8bf hi) {
  return __builtin_shufflevector(lo, hi, 0,1,2,3,4,5,6,7,8,9,10,11,12,13,14,15);
}
__device__ __forceinline__ v8f wmma_bf16(v16bf a, v16bf b, v8f c) {
  return __builtin_amdgcn_wmma_f32_16x16x32_bf16(false, a, false, b, (short)0, c, false, false);
}

// ---------------- CDNA5 async global->LDS copy (ASYNCcnt path), with fallback
#if __has_builtin(__builtin_amdgcn_global_load_async_to_lds_b128)
#define USE_ASYNC 1
#else
#define USE_ASYNC 0
#endif

__device__ __forceinline__ void cp16_g2l(unsigned short* lds_dst, const unsigned short* gsrc) {
#if USE_ASYNC
  __builtin_amdgcn_global_load_async_to_lds_b128(
      (__attribute__((address_space(1))) v4i*)gsrc,
      (__attribute__((address_space(3))) v4i*)lds_dst, 0, 0);
#else
  *reinterpret_cast<uint4*>(lds_dst) = *reinterpret_cast<const uint4*>(gsrc);
#endif
}
__device__ __forceinline__ void wait_async_all() {
#if USE_ASYNC
#if __has_builtin(__builtin_amdgcn_s_wait_asynccnt)
  __builtin_amdgcn_s_wait_asynccnt(0);
#else
  asm volatile("s_wait_asynccnt 0" ::: "memory");
#endif
#endif
}

// ---------------------------------------------------------------- convert
__global__ void cvt_f32_bf16(const float* __restrict__ in,
                             unsigned short* __restrict__ out, int n) {
  int i = blockIdx.x * blockDim.x + threadIdx.x;
  int stride = gridDim.x * blockDim.x;
  for (; i < n; i += stride) out[i] = f2bf(in[i]);
}

// ----------------------------------------------------- LDS-tiled WMMA GEMM
// C[M,N] = A[M,K] @ Bw[N,K]^T (+bias). 128x128 block tile, 8 waves (4x2),
// each wave 32x64 = 2x4 WMMA subtiles. K staged 32 wide, double-buffered LDS
// filled with async global->LDS b128 copies.
// MODE 0: epilogue scatters to Q[B,H,S,Dh] (pre-scaled), K[B,H,S,Dh], Vt[B,H,Dh,S]
// MODE 1: epilogue stores fp32 to Out[M, D_MODEL]
template <int MODE>
__global__ __launch_bounds__(256) void gemm_tiled(
    const unsigned short* __restrict__ A,   // [M, 2048] bf16
    const unsigned short* __restrict__ Bw,  // [Ndim, 2048] bf16
    const float* __restrict__ bias,         // [Ndim]
    int Ndim,
    unsigned short* __restrict__ Q,
    unsigned short* __restrict__ Ko,
    unsigned short* __restrict__ Vt,
    float* __restrict__ Out) {
  __shared__ unsigned short As[2][128][32];   // 16 KB
  __shared__ unsigned short Bs[2][128][32];   // 16 KB
  const int NB = Ndim / 128;
  const int mb = blockIdx.x / NB, nb = blockIdx.x % NB;
  const int t = threadIdx.x;
  const int lane = t & 31, w = t >> 5;
  const int hsel = lane >> 4, ln = lane & 15;
  const int wr = w >> 1, wc = w & 1;          // 4 x 2 wave grid
  const size_t mbase = (size_t)mb * 128, nbase = (size_t)nb * 128;

  const int chunk = t & 3;                    // 16B chunk within 32-half row
  const int row0  = t >> 2;                   // 0..63

  auto stage = [&](int buf, int k0) {
#pragma unroll
    for (int rr = 0; rr < 2; ++rr) {
      int row = row0 + rr * 64;
      cp16_g2l(&As[buf][row][chunk * 8],
               A  + (mbase + row) * D_MODEL + k0 + chunk * 8);
      cp16_g2l(&Bs[buf][row][chunk * 8],
               Bw + (nbase + row) * D_MODEL + k0 + chunk * 8);
    }
  };

  v8f zero = {};
  v8f Cacc[2][4];
#pragma unroll
  for (int su = 0; su < 2; ++su)
#pragma unroll
    for (int sv = 0; sv < 4; ++sv) Cacc[su][sv] = zero;

  stage(0, 0);
  for (int k0 = 0, it = 0; k0 < D_MODEL; k0 += 32, ++it) {
    const int cur = it & 1;
    wait_async_all();       // my async writes to LDS have landed
    __syncthreads();        // everyone's have landed; prev buffer fully read
    if (k0 + 32 < D_MODEL) stage(cur ^ 1, k0 + 32);

    v16bf a[2], b[4];
#pragma unroll
    for (int su = 0; su < 2; ++su) {
      const unsigned short* p = &As[cur][wr * 32 + su * 16 + ln][0];
      a[su] = cat8(ld8(p + hsel * 8), ld8(p + 16 + hsel * 8));
    }
#pragma unroll
    for (int sv = 0; sv < 4; ++sv) {
      const unsigned short* p = &Bs[cur][wc * 64 + sv * 16 + ln][0];
      b[sv] = cat8(ld8(p + hsel * 8), ld8(p + 16 + hsel * 8));
    }
#pragma unroll
    for (int su = 0; su < 2; ++su)
#pragma unroll
      for (int sv = 0; sv < 4; ++sv)
        Cacc[su][sv] = wmma_bf16(a[su], b[sv], Cacc[su][sv]);
  }

#pragma unroll
  for (int su = 0; su < 2; ++su) {
#pragma unroll
    for (int sv = 0; sv < 4; ++sv) {
      const int n = (int)nbase + wc * 64 + sv * 16 + ln;
      const float bv = bias[n];
#pragma unroll
      for (int r = 0; r < 8; ++r) {
        const int m = (int)mbase + wr * 32 + su * 16 + r + hsel * 8;
        const float v = Cacc[su][sv][r] + bv;
        if (MODE == 0) {
          const int sec = n / D_MODEL, d = n % D_MODEL;
          const int h = d / D_HEAD, dh = d % D_HEAD;
          const int bb = m / SEQ, s = m % SEQ;
          if (sec == 0)
            Q [((size_t)(bb * NUM_HEADS + h) * SEQ + s) * D_HEAD + dh] = f2bf(v * SCALE_Q);
          else if (sec == 1)
            Ko[((size_t)(bb * NUM_HEADS + h) * SEQ + s) * D_HEAD + dh] = f2bf(v);
          else
            Vt[((size_t)(bb * NUM_HEADS + h) * D_HEAD + dh) * SEQ + s] = f2bf(v);
        } else {
          Out[(size_t)m * D_MODEL + n] = v;
        }
      }
    }
  }
}

// ---------------------------------------------------------- flash attention
// One wave per 16-query tile; 32-key tiles with online softmax; ALiBi + causal.
__global__ __launch_bounds__(256) void flash_attn(
    const unsigned short* __restrict__ Q,
    const unsigned short* __restrict__ K,
    const unsigned short* __restrict__ Vt,
    const float* __restrict__ slopes,
    unsigned short* __restrict__ Oa) {       // attended bf16 [M_TOTAL, D_MODEL]
  __shared__ unsigned short pl[8][16][32];   // per-wave P tile (C-layout -> A-layout)
  const int lane = threadIdx.x & 31;
  const int w    = threadIdx.x >> 5;
  const int wid  = (blockIdx.x * blockDim.x + threadIdx.x) >> 5;
  const int QT = SEQ / 16;                   // 128 query tiles per (b,h)
  const int qt = wid % QT;
  const int bh = wid / QT;                   // 0..63
  const int h  = bh % NUM_HEADS;
  const int q0 = qt * 16;
  const float slope = slopes[h];
  const int hsel = lane >> 4, ln = lane & 15;

  const unsigned short* Qb = Q  + (size_t)bh * SEQ * D_HEAD;
  const unsigned short* Kb = K  + (size_t)bh * SEQ * D_HEAD;
  const unsigned short* Vb = Vt + (size_t)bh * D_HEAD * SEQ;

  v16bf aq[4];
#pragma unroll
  for (int cdh = 0; cdh < 4; ++cdh) {
    const unsigned short* qrow = Qb + (size_t)(q0 + ln) * D_HEAD + cdh * 32;
    aq[cdh] = cat8(ld8(qrow + hsel * 8), ld8(qrow + 16 + hsel * 8));
  }

  v8f zero = {};
  v8f O[8];
#pragma unroll
  for (int d = 0; d < 8; ++d) O[d] = zero;
  float vm[8], vl[8];
#pragma unroll
  for (int r = 0; r < 8; ++r) { vm[r] = -1e30f; vl[r] = 0.f; }

  const int kend = q0 + 15;
  for (int kt = 0; kt <= kend; kt += 32) {
    v8f c0 = zero, c1 = zero;
#pragma unroll
    for (int cdh = 0; cdh < 4; ++cdh) {
      const unsigned short* k0r = Kb + (size_t)(kt + ln) * D_HEAD + cdh * 32;
      const unsigned short* k1r = Kb + (size_t)(kt + 16 + ln) * D_HEAD + cdh * 32;
      c0 = wmma_bf16(aq[cdh], cat8(ld8(k0r + hsel * 8), ld8(k0r + 16 + hsel * 8)), c0);
      c1 = wmma_bf16(aq[cdh], cat8(ld8(k1r + hsel * 8), ld8(k1r + 16 + hsel * 8)), c1);
    }
    const int key0 = kt + ln;
    const int key1 = kt + 16 + ln;
#pragma unroll
    for (int r = 0; r < 8; ++r) {
      const int m    = r + hsel * 8;
      const int qpos = q0 + m;
      float x0 = (key0 <= qpos) ? (c0[r] + slope * (float)(key0 - qpos)) * LOG2E : -1e30f;
      float x1 = (key1 <= qpos) ? (c1[r] + slope * (float)(key1 - qpos)) * LOG2E : -1e30f;
      float loc = fmaxf(x0, x1);
      loc = fmaxf(loc, __shfl_xor(loc, 1, 16));
      loc = fmaxf(loc, __shfl_xor(loc, 2, 16));
      loc = fmaxf(loc, __shfl_xor(loc, 4, 16));
      loc = fmaxf(loc, __shfl_xor(loc, 8, 16));
      float mnew = fmaxf(vm[r], loc);
      float p0 = exp2f(x0 - mnew);
      float p1 = exp2f(x1 - mnew);
      float ps = p0 + p1;
      ps += __shfl_xor(ps, 1, 16);
      ps += __shfl_xor(ps, 2, 16);
      ps += __shfl_xor(ps, 4, 16);
      ps += __shfl_xor(ps, 8, 16);
      float corr = exp2f(vm[r] - mnew);
      vl[r] = vl[r] * corr + ps;
      vm[r] = mnew;
#pragma unroll
      for (int d = 0; d < 8; ++d) O[d][r] *= corr;
      pl[w][m][ln]      = f2bf(p0);
      pl[w][m][16 + ln] = f2bf(p1);
    }
    const unsigned short* prow = &pl[w][ln][0];
    v16bf pa = cat8(ld8(prow + hsel * 8), ld8(prow + 16 + hsel * 8));
#pragma unroll
    for (int d = 0; d < 8; ++d) {
      const unsigned short* vrow = Vb + (size_t)(d * 16 + ln) * SEQ + kt;
      O[d] = wmma_bf16(pa, cat8(ld8(vrow + hsel * 8), ld8(vrow + 16 + hsel * 8)), O[d]);
    }
  }

  float inv[8];
#pragma unroll
  for (int r = 0; r < 8; ++r) inv[r] = 1.0f / fmaxf(vl[r], 1e-20f);
  const int bb = bh / NUM_HEADS;
#pragma unroll
  for (int d = 0; d < 8; ++d) {
#pragma unroll
    for (int r = 0; r < 8; ++r) {
      int m = r + hsel * 8;
      int s = q0 + m;
      Oa[((size_t)(bb * SEQ + s)) * D_MODEL + h * D_HEAD + d * 16 + ln] =
          f2bf(O[d][r] * inv[r]);
    }
  }
}

// ---------------------------------------------------------------- launcher
extern "C" void kernel_launch(void* const* d_in, const int* in_sizes, int n_in,
                              void* d_out, int out_size, void* d_ws, size_t ws_size,
                              hipStream_t stream) {
  (void)in_sizes; (void)n_in; (void)out_size; (void)ws_size;
  const float* qkv    = (const float*)d_in[0];
  const float* win    = (const float*)d_in[1];
  const float* bin    = (const float*)d_in[2];
  const float* wout   = (const float*)d_in[3];
  const float* bout   = (const float*)d_in[4];
  const float* slopes = (const float*)d_in[5];
  float* out = (float*)d_out;

  char* ws = (char*)d_ws;
  size_t off = 0;
  auto take = [&](size_t bytes) -> void* {
    void* p = ws + off;
    off = (off + bytes + 255) & ~(size_t)255;
    return p;
  };
  unsigned short* qkv_bf  = (unsigned short*)take((size_t)M_TOTAL * D_MODEL * 2);
  unsigned short* win_bf  = (unsigned short*)take((size_t)N_PROJ  * D_MODEL * 2);
  unsigned short* wout_bf = (unsigned short*)take((size_t)D_MODEL * D_MODEL * 2);
  unsigned short* Qb      = (unsigned short*)take((size_t)M_TOTAL * D_MODEL * 2);
  unsigned short* Kb      = (unsigned short*)take((size_t)M_TOTAL * D_MODEL * 2);
  unsigned short* Vb      = (unsigned short*)take((size_t)M_TOTAL * D_MODEL * 2);
  unsigned short* Ab      = (unsigned short*)take((size_t)M_TOTAL * D_MODEL * 2);

  cvt_f32_bf16<<<4096, 256, 0, stream>>>(qkv,  qkv_bf,  M_TOTAL * D_MODEL);
  cvt_f32_bf16<<<4096, 256, 0, stream>>>(win,  win_bf,  N_PROJ  * D_MODEL);
  cvt_f32_bf16<<<4096, 256, 0, stream>>>(wout, wout_bf, D_MODEL * D_MODEL);

  // in-proj: M=8192 (64 tiles) x N=6144 (48 tiles)
  gemm_tiled<0><<<64 * 48, 256, 0, stream>>>(qkv_bf, win_bf, bin, N_PROJ,
                                             Qb, Kb, Vb, nullptr);
  flash_attn<<<(64 * 128) / 8, 256, 0, stream>>>(Qb, Kb, Vb, slopes, Ab);
  // out-proj: M=8192 (64 tiles) x N=2048 (16 tiles)
  gemm_tiled<1><<<64 * 16, 256, 0, stream>>>(Ab, wout_bf, bout, D_MODEL,
                                             nullptr, nullptr, nullptr, out);
}
